// GAE_10685878632448
// MI455X (gfx1250) — matrix operand order repaired and checked by reading, not compiled, not run on hardware
//
#include <hip/hip_runtime.h>

typedef _Float16 v16h __attribute__((ext_vector_type(16)));
typedef _Float16 v8h  __attribute__((ext_vector_type(8)));
typedef float    v8f  __attribute__((ext_vector_type(8)));
typedef float    v4f  __attribute__((ext_vector_type(4)));

// ---------------- degree / norm ----------------

__global__ void k_fill1(float* deg, int n) {
    int i = blockIdx.x * blockDim.x + threadIdx.x;
    if (i < n) deg[i] = 1.0f;   // self-loop contributes 1 to every node's degree
}

__global__ void k_deg(const int* __restrict__ ei, float* deg, int E) {
    int e = blockIdx.x * blockDim.x + threadIdx.x;
    if (e >= E) return;
    int d = ei[E + e];          // row 1 = dst
    __hip_atomic_fetch_add(deg + d, 1.0f, __ATOMIC_RELAXED, __HIP_MEMORY_SCOPE_AGENT);
}

__global__ void k_rsqrt(float* deg, int n) {
    int i = blockIdx.x * blockDim.x + threadIdx.x;
    if (i < n) deg[i] = rsqrtf(deg[i]);   // deg >= 1 always (self-loops)
}

// ---------------- precision conversion / weight pre-swizzle ----------------

__global__ void k_f32_to_f16(const float* __restrict__ s, _Float16* __restrict__ d, int n) {
    int i = blockIdx.x * blockDim.x + threadIdx.x;
    if (i < n) d[i] = (_Float16)s[i];
}

// Pre-swizzle W[K][N] (row-major f32) into WMMA B-fragment order:
// tile (nt,kt) is 32x16, laid out so lane L owns 16 contiguous halves:
//   k = kt*32 + (L>=16 ? 16 : 0) + h,  n = nt*16 + (L & 15)
__global__ void k_prep_w(const float* __restrict__ W, _Float16* __restrict__ Bp, int K, int N) {
    int idx = blockIdx.x * blockDim.x + threadIdx.x;
    if (idx >= K * N) return;
    int KT = K >> 5;
    int h    =  idx        & 15;
    int lane = (idx >> 4)  & 31;
    int kt   = (idx >> 9) % KT;
    int nt   = (idx >> 9) / KT;
    int k = kt * 32 + ((lane >= 16) ? 16 : 0) + h;
    int n = nt * 16 + (lane & 15);
    Bp[idx] = (_Float16)W[(long)k * N + n];
}

// ---------------- WMMA GEMM: C[M,N] = A[M,K] * B[K,N] ----------------
// A: f16 row-major. Bp: f16 in fragment order (see k_prep_w). C: f32 row-major.
// One wave computes a 16x64 output strip (4 accumulators, A-frag reused 4x).
__global__ void k_gemm_wmma(const _Float16* __restrict__ A,
                            const _Float16* __restrict__ Bp,
                            float* __restrict__ C, int M, int K, int N) {
    int wave = (blockIdx.x * blockDim.x + threadIdx.x) >> 5;
    int lane = threadIdx.x & 31;
    int ngrp = N >> 6;                 // 64-column groups
    int mt = wave / ngrp;
    int ng = wave % ngrp;
    if (mt >= (M >> 4)) return;        // wave-uniform: EXEC stays all-ones
    int KT = K >> 5;

    v8f acc0 = {}, acc1 = {}, acc2 = {}, acc3 = {};

    int m = mt * 16 + (lane & 15);
    int kbaseA = (lane >= 16) ? 16 : 0;
    const _Float16* arow = A + (long)m * K;

    for (int kt = 0; kt < KT; ++kt) {
        int k0 = kt * 32 + kbaseA;
        v8h alo = *(const v8h*)(arow + k0);        // K {0..7}  (+16 for hi lanes)
        v8h ahi = *(const v8h*)(arow + k0 + 8);    // K {8..15} (+16 for hi lanes)
        v16h a = __builtin_shufflevector(alo, ahi,
                 0,1,2,3,4,5,6,7,8,9,10,11,12,13,14,15);

        const _Float16* bbase = Bp + (((long)ng * 4 * KT + kt) * 512) + lane * 16;
#pragma unroll
        for (int t = 0; t < 4; ++t) {
            const _Float16* bt = bbase + (long)t * KT * 512;
            v8h blo = *(const v8h*)(bt);
            v8h bhi = *(const v8h*)(bt + 8);
            v16h b = __builtin_shufflevector(blo, bhi,
                     0,1,2,3,4,5,6,7,8,9,10,11,12,13,14,15);
            v8f c = (t == 0) ? acc0 : (t == 1) ? acc1 : (t == 2) ? acc2 : acc3;
            c = __builtin_amdgcn_wmma_f32_16x16x32_f16(
                    false, a, false, b, (short)0, c, false, false);
            if (t == 0) acc0 = c; else if (t == 1) acc1 = c;
            else if (t == 2) acc2 = c; else acc3 = c;
        }
    }

    // C/D layout: VGPR r -> row mt*16 + r + (lane>=16)*8, col = lane&15
    int mrow = mt * 16 + ((lane >= 16) ? 8 : 0);
    int ncol = (lane & 15);
#pragma unroll
    for (int t = 0; t < 4; ++t) {
        v8f c = (t == 0) ? acc0 : (t == 1) ? acc1 : (t == 2) ? acc2 : acc3;
        int n = (ng * 4 + t) * 16 + ncol;
#pragma unroll
        for (int r = 0; r < 8; ++r)
            C[(long)(mrow + r) * N + n] = c[r];
    }
}

// ---------------- aggregation ----------------

// Self-loop term without atomics: AGG[i,:] = H[i,:] * dinv[i]^2
__global__ void k_selfloop_init(const float* __restrict__ H, const float* __restrict__ dinv,
                                float* __restrict__ AGG, int n, int C) {
    int idx = blockIdx.x * blockDim.x + threadIdx.x;
    if (idx >= n * C) return;
    int i = idx / C;
    float w = dinv[i];
    AGG[idx] = H[idx] * (w * w);
}

// One wave per edge: gather H[src], scale, atomic scatter-add into AGG[dst]
__global__ void k_edge_agg(const int* __restrict__ ei, const float* __restrict__ H,
                           const float* __restrict__ dinv, float* AGG, int E, int C) {
    int wave = (blockIdx.x * blockDim.x + threadIdx.x) >> 5;
    int lane = threadIdx.x & 31;
    if (wave >= E) return;
    int s = ei[wave];
    int d = ei[E + wave];
    float nrm = dinv[s] * dinv[d];
    int per = C >> 5;                  // 8 (C=256) or 4 (C=128) channels per lane
    const float* hs = H + (long)s * C + lane * per;
    float* ag = AGG + (long)d * C + lane * per;
    for (int c = 0; c < per; c += 4) {
        v4f v = *(const v4f*)(hs + c);
#pragma unroll
        for (int j = 0; j < 4; ++j)
            __hip_atomic_fetch_add(ag + c + j, v[j] * nrm,
                                   __ATOMIC_RELAXED, __HIP_MEMORY_SCOPE_AGENT);
    }
}

// bias + relu, convert to f16 for the next WMMA GEMM
__global__ void k_bias_relu_f16(const float* __restrict__ AGG, const float* __restrict__ b,
                                _Float16* __restrict__ Z, int n, int C) {
    int idx = blockIdx.x * blockDim.x + threadIdx.x;
    if (idx >= n * C) return;
    float v = AGG[idx] + b[idx & (C - 1)];
    Z[idx] = (_Float16)fmaxf(v, 0.0f);
}

// bias only (in place), f32 output for decode
__global__ void k_bias(float* AGG, const float* __restrict__ b, int n, int C) {
    int idx = blockIdx.x * blockDim.x + threadIdx.x;
    if (idx >= n * C) return;
    AGG[idx] = AGG[idx] + b[idx & (C - 1)];
}

// ---------------- decode: out[e] = dot(Z[src], Z[dst]) over 128 ch ----------------

__global__ void k_decode(const int* __restrict__ ei, const float* __restrict__ Z,
                         float* __restrict__ out, int E) {
    int wave = (blockIdx.x * blockDim.x + threadIdx.x) >> 5;
    int lane = threadIdx.x & 31;
    if (wave >= E) return;
    int a = ei[wave];
    int b = ei[E + wave];
    v4f za = *(const v4f*)(Z + (long)a * 128 + lane * 4);
    v4f zb = *(const v4f*)(Z + (long)b * 128 + lane * 4);
    float p = za[0] * zb[0] + za[1] * zb[1] + za[2] * zb[2] + za[3] * zb[3];
#pragma unroll
    for (int off = 16; off > 0; off >>= 1)
        p += __shfl_down(p, off, 32);
    if (lane == 0) out[wave] = p;
}

// ---------------- launch ----------------

static inline int cdiv(long a, long b) { return (int)((a + b - 1) / b); }

extern "C" void kernel_launch(void* const* d_in, const int* in_sizes, int n_in,
                              void* d_out, int out_size, void* d_ws, size_t ws_size,
                              hipStream_t stream) {
    const int IN_CH = 256, HID = 256, OUT_CH = 128;
    const int N = in_sizes[0] / IN_CH;   // 50000
    const int E = in_sizes[1] / 2;       // 1.6M

    const float* x  = (const float*)d_in[0];
    const int*   ei = (const int*)d_in[1];
    const float* W1 = (const float*)d_in[2];
    const float* b1 = (const float*)d_in[3];
    const float* W2 = (const float*)d_in[4];
    const float* b2 = (const float*)d_in[5];
    float* out = (float*)d_out;

    // workspace carve-out
    size_t off = 0;
    auto carve = [&](size_t bytes) {
        size_t o = off; off += (bytes + 255) & ~(size_t)255; return o;
    };
    char* ws = (char*)d_ws;
    float*    dinv = (float*)   (ws + carve((size_t)N * 4));
    _Float16* Xh   = (_Float16*)(ws + carve((size_t)N * HID * 2));   // reused as Z1h
    _Float16* W1p  = (_Float16*)(ws + carve((size_t)IN_CH * HID * 2));
    _Float16* W2p  = (_Float16*)(ws + carve((size_t)HID * OUT_CH * 2));
    float*    Hb   = (float*)   (ws + carve((size_t)N * HID * 4));   // H1, reused as H2
    float*    AGG  = (float*)   (ws + carve((size_t)N * HID * 4));   // AGG1, reused AGG2/Z2
    (void)ws_size;

    const int T = 256;

    // degrees -> dinv
    k_fill1 <<<cdiv(N, T), T, 0, stream>>>(dinv, N);
    k_deg   <<<cdiv(E, T), T, 0, stream>>>(ei, dinv, E);
    k_rsqrt <<<cdiv(N, T), T, 0, stream>>>(dinv, N);

    // f16 conversions + weight fragment pre-swizzle
    k_f32_to_f16 <<<cdiv((long)N * IN_CH, T), T, 0, stream>>>(x, Xh, N * IN_CH);
    k_prep_w     <<<cdiv(IN_CH * HID, T), T, 0, stream>>>(W1, W1p, IN_CH, HID);
    k_prep_w     <<<cdiv(HID * OUT_CH, T), T, 0, stream>>>(W2, W2p, HID, OUT_CH);

    // layer 1: H1 = X @ W1 ; aggregate ; relu(. + b1) -> f16
    {
        long waves = (long)(N / 16) * (HID / 64);
        k_gemm_wmma <<<cdiv(waves * 32, T), T, 0, stream>>>(Xh, W1p, Hb, N, IN_CH, HID);
    }
    k_selfloop_init <<<cdiv((long)N * HID, T), T, 0, stream>>>(Hb, dinv, AGG, N, HID);
    k_edge_agg      <<<cdiv((long)E * 32, T), T, 0, stream>>>(ei, Hb, dinv, AGG, E, HID);
    k_bias_relu_f16 <<<cdiv((long)N * HID, T), T, 0, stream>>>(AGG, b1, Xh, N, HID);

    // layer 2: H2 = Z1 @ W2 ; aggregate ; + b2
    {
        long waves = (long)(N / 16) * (OUT_CH / 64);
        k_gemm_wmma <<<cdiv(waves * 32, T), T, 0, stream>>>(Xh, W2p, Hb, N, HID, OUT_CH);
    }
    k_selfloop_init <<<cdiv((long)N * OUT_CH, T), T, 0, stream>>>(Hb, dinv, AGG, N, OUT_CH);
    k_edge_agg      <<<cdiv((long)E * 32, T), T, 0, stream>>>(ei, Hb, dinv, AGG, E, OUT_CH);
    k_bias          <<<cdiv((long)N * OUT_CH, T), T, 0, stream>>>(AGG, b2, N, OUT_CH);

    // decode
    k_decode <<<cdiv((long)E * 32, T), T, 0, stream>>>(ei, AGG, out, E);
}
